// ModuleGroup_7009386627576
// MI455X (gfx1250) — matrix operand order repaired and checked by reading, not compiled, not run on hardware
//
#include <hip/hip_runtime.h>
#include <hip/hip_bf16.h>

// ---------------------------------------------------------------------------
// MoE expert MLP (top-2, capacity 1024) for MI455X (gfx1250, wave32, WMMA).
// bf16 tensor-core GEMMs, f32 accumulate. Double-buffered LDS staging;
// A-tiles streamed with GLOBAL_LOAD_ASYNC_TO_LDS_B128 (ASYNCcnt), B-tiles
// converted f32->bf16 in-register (weights read once from HBM; per-expert
// weight set is L2-resident at 16 MB per GEMM).
// ---------------------------------------------------------------------------

constexpr int Tn   = 4096;
constexpr int En   = 8;
constexpr int Dn   = 1024;
constexpr int Fn   = 4096;
constexpr int CAPn = 1024;

typedef __attribute__((ext_vector_type(16))) __bf16 v16bf;
typedef __attribute__((ext_vector_type(8)))  float  v8f;

// GEMM tiling
constexpr int BM = 128;
constexpr int BN = 128;
constexpr int BK = 32;
constexpr int LDA = BK + 8;   // ushort stride of A tile in LDS (padded)
constexpr int LDB = BN + 8;   // ushort stride of B tile in LDS (padded)

union AFrag { v16bf v; unsigned int w[8]; unsigned short u[16]; };
union BFrag { v16bf v; uint4 q[2];        unsigned short u[16]; };
union AccT  { v8f   v; float f[8]; };

__device__ __forceinline__ unsigned short f2bf(float f) {
    union { float f; unsigned int u; } c; c.f = f;
    return (unsigned short)((c.u + 0x7FFFu + ((c.u >> 16) & 1u)) >> 16);  // RNE
}

// ---------------------------------------------------------------------------
// Async global->LDS (CDNA5): per-lane 16B copy, tracked by ASYNCcnt.
// Builtin signature (from hipcc diagnostic): arg0 = AS1 pointer to
// int __attribute__((vector_size(16))), arg1 = LDS pointer, + imm offset/cpol.
// AS pointers formed via integer round-trip (flat aperture: LDS offset
// = addr[31:0]; global addresses are identity in the global aperture).
// ---------------------------------------------------------------------------
#if defined(__gfx1250__) && __has_builtin(__builtin_amdgcn_global_load_async_to_lds_b128)
#define USE_ASYNC_LDS 1
#endif

typedef int v4i_gcc __attribute__((vector_size(16)));
typedef __attribute__((address_space(1))) v4i_gcc* as1_v4i_ptr;
typedef __attribute__((address_space(3))) v4i_gcc* as3_v4i_ptr;

__device__ __forceinline__ void copy_b128_to_lds(const void* g, void* lds) {
#ifdef USE_ASYNC_LDS
    __builtin_amdgcn_global_load_async_to_lds_b128(
        (as1_v4i_ptr)(unsigned long long)g,
        (as3_v4i_ptr)(unsigned int)(unsigned long long)lds,
        0, 0);
#else
    *reinterpret_cast<uint4*>(lds) = *reinterpret_cast<const uint4*>(g);
#endif
}

__device__ __forceinline__ void async_wait0() {
#ifdef USE_ASYNC_LDS
#if __has_builtin(__builtin_amdgcn_s_wait_asynccnt)
    __builtin_amdgcn_s_wait_asynccnt(0);
#else
    asm volatile("s_wait_asynccnt 0x0" ::: "memory");
#endif
#endif
}

// ---------------------------------------------------------------------------
// Kernel 1: zero-init output, counters, slot tables.
// ---------------------------------------------------------------------------
__global__ __launch_bounds__(256) void moe_init(int* __restrict__ cnt,
                                                int* __restrict__ token_idx,
                                                float* __restrict__ token_w,
                                                float* __restrict__ y) {
    int i = blockIdx.x * blockDim.x + threadIdx.x;
    if (i < En) cnt[i] = 0;
    if (i < En * CAPn) { token_idx[i] = -1; token_w[i] = 0.0f; }
    if (i < Tn * Dn) y[i] = 0.0f;
}

// ---------------------------------------------------------------------------
// Kernel 2: routing -> per-expert slot lists via atomic append (capacity CAP).
// ---------------------------------------------------------------------------
__global__ __launch_bounds__(256) void moe_route(const unsigned char* __restrict__ mask,
                                                 const float* __restrict__ rweight,
                                                 int* __restrict__ cnt,
                                                 int* __restrict__ token_idx,
                                                 float* __restrict__ token_w) {
    int t = blockIdx.x * blockDim.x + threadIdx.x;
    if (t >= Tn) return;
    #pragma unroll
    for (int e = 0; e < En; ++e) {
        if (mask[t * En + e]) {
            int pos = atomicAdd(&cnt[e], 1);
            if (pos < CAPn) {
                token_idx[e * CAPn + pos] = t;
                token_w[e * CAPn + pos]   = rweight[t * En + e];
            }
        }
    }
}

// ---------------------------------------------------------------------------
// Kernel 3: dispatch gather, f32 -> bf16 (zero-fill unused slots).
// ---------------------------------------------------------------------------
__global__ __launch_bounds__(256) void moe_gather(const float* __restrict__ x,
                                                  const int* __restrict__ token_idx,
                                                  unsigned short* __restrict__ xs) {
    int ec = blockIdx.x;                 // e*CAP + c
    int t  = token_idx[ec];
    size_t base = (size_t)ec * Dn;
    int d = threadIdx.x * 4;
    if (t >= 0) {
        const float4 f = *reinterpret_cast<const float4*>(&x[(size_t)t * Dn + d]);
        ushort4 o;
        o.x = f2bf(f.x); o.y = f2bf(f.y); o.z = f2bf(f.z); o.w = f2bf(f.w);
        *reinterpret_cast<ushort4*>(&xs[base + d]) = o;
    } else {
        *reinterpret_cast<ushort4*>(&xs[base + d]) = make_ushort4(0, 0, 0, 0);
    }
}

// ---------------------------------------------------------------------------
// Shared GEMM core: C[BMxBN] += A[BMxKDIM] * B[KDIMxBN].
// A bf16 in global (async-streamed to LDS), B f32 in global (converted to
// bf16 in-register during staging). Double-buffered LDS: stage k+1 overlaps
// WMMA on k. 256 threads = 8 waves (2x4); each wave owns 64x32 of C
// (4x2 accumulator tiles; v_wmma_f32_16x16x32_bf16).
// ---------------------------------------------------------------------------
template <int KDIM, int NDIM>
__device__ __forceinline__ void gemm_core(const unsigned short* __restrict__ Amat,
                                          const float* __restrict__ Bmat,
                                          int m0, int n0,
                                          unsigned short (&sA)[2][BM * LDA],
                                          unsigned short (&sB)[2][BK * LDB],
                                          AccT acc[4][2]) {
    const int tid  = threadIdx.x;
    const int lane = tid & 31;
    const int wave = tid >> 5;
    const int wm   = (wave >> 2) * 64;   // wave row offset within BM
    const int wn   = (wave & 3) * 32;    // wave col offset within BN
    const int l    = lane & 15;
    const int hi   = lane >> 4;

    const int arow  = tid >> 1, ahalf = tid & 1;         // A: 16 bf16 per thread
    const int brow  = tid >> 3, bnb   = (tid & 7) * 16;  // B: 16 f32 per thread

    auto stageA = [&](int buf, int k0) {
        const unsigned short* g = &Amat[(size_t)(m0 + arow) * KDIM + k0 + ahalf * 16];
        unsigned short*       d = &sA[buf][arow * LDA + ahalf * 16];
        copy_b128_to_lds(g,     d);       // 8 bf16
        copy_b128_to_lds(g + 8, d + 8);   // 8 bf16
    };
    auto stageB = [&](int buf, int k0) {
        const float4* src = reinterpret_cast<const float4*>(
            &Bmat[(size_t)(k0 + brow) * NDIM + n0 + bnb]);
        unsigned short tmp[16];
        #pragma unroll
        for (int q = 0; q < 4; ++q) {
            float4 f = src[q];
            tmp[q * 4 + 0] = f2bf(f.x);
            tmp[q * 4 + 1] = f2bf(f.y);
            tmp[q * 4 + 2] = f2bf(f.z);
            tmp[q * 4 + 3] = f2bf(f.w);
        }
        uint4* dst = reinterpret_cast<uint4*>(&sB[buf][brow * LDB + bnb]);
        dst[0] = *reinterpret_cast<uint4*>(&tmp[0]);
        dst[1] = *reinterpret_cast<uint4*>(&tmp[8]);
        if (k0 + BK < KDIM)   // -> global_prefetch_b8 for the tile after next
            __builtin_prefetch(&Bmat[(size_t)(k0 + BK + brow) * NDIM + n0 + bnb], 0, 1);
    };

    constexpr int nk = KDIM / BK;

    stageA(0, 0);
    stageB(0, 0);
    async_wait0();
    __syncthreads();

    for (int kt = 0; kt < nk; ++kt) {
        const int cur = kt & 1;
        if (kt + 1 < nk) {                 // overlap staging with compute
            stageA(cur ^ 1, (kt + 1) * BK);
            stageB(cur ^ 1, (kt + 1) * BK);
        }

        // --- fragments per CDNA5 WMMA VGPR layouts + 8 WMMAs ---
        AFrag af[4];
        BFrag bq[2];
        #pragma unroll
        for (int i = 0; i < 4; ++i) {
            int r = wm + i * 16 + l;       // A row (M) for this lane
            #pragma unroll
            for (int vv = 0; vv < 8; ++vv) {
                int kb = ((vv < 4) ? 2 * vv : 16 + 2 * (vv - 4)) + hi * 8;
                af[i].w[vv] = *reinterpret_cast<const unsigned int*>(&sA[cur][r * LDA + kb]);
            }
        }
        #pragma unroll
        for (int j = 0; j < 2; ++j) {
            #pragma unroll
            for (int s = 0; s < 16; ++s) {
                int kk = ((s < 8) ? s : 8 + s) + hi * 8;
                bq[j].u[s] = sB[cur][kk * LDB + wn + j * 16 + l];
            }
        }
        #pragma unroll
        for (int i = 0; i < 4; ++i)
            #pragma unroll
            for (int j = 0; j < 2; ++j)
                acc[i][j].v = __builtin_amdgcn_wmma_f32_16x16x32_bf16(
                    false, af[i].v, false, bq[j].v, (short)0, acc[i][j].v, false, false);

        async_wait0();                     // async A writes to buf^1 landed
        __syncthreads();
    }
}

// ---------------------------------------------------------------------------
// Kernel 4: h = gelu(xs @ w1[e] + b1[e]),  (E, C, F) bf16
// ---------------------------------------------------------------------------
__global__ __launch_bounds__(256) void moe_gemm1(const unsigned short* __restrict__ xs,
                                                 const float* __restrict__ w1,
                                                 const float* __restrict__ b1,
                                                 unsigned short* __restrict__ hbuf) {
    __shared__ unsigned short sA[2][BM * LDA];
    __shared__ unsigned short sB[2][BK * LDB];

    const int e  = blockIdx.z;
    const int n0 = blockIdx.x * BN;
    const int m0 = blockIdx.y * BM;
    const int lane = threadIdx.x & 31;
    const int wave = threadIdx.x >> 5;
    const int wm = (wave >> 2) * 64;
    const int wn = (wave & 3) * 32;
    const int l  = lane & 15;
    const int hi = lane >> 4;

    AccT acc[4][2];
    #pragma unroll
    for (int i = 0; i < 4; ++i)
        #pragma unroll
        for (int j = 0; j < 2; ++j)
            acc[i][j].v = (v8f){0.f, 0.f, 0.f, 0.f, 0.f, 0.f, 0.f, 0.f};

    const unsigned short* Ae = xs + (size_t)e * CAPn * Dn;
    const float* Be = w1 + (size_t)e * Dn * Fn;
    gemm_core<Dn, Fn>(Ae, Be, m0, n0, sA, sB, acc);

    // epilogue: bias + GELU, store bf16
    #pragma unroll
    for (int i = 0; i < 4; ++i) {
        #pragma unroll
        for (int j = 0; j < 2; ++j) {
            int col = n0 + wn + j * 16 + l;
            float bias = b1[e * Fn + col];
            #pragma unroll
            for (int r = 0; r < 8; ++r) {
                int row = m0 + wm + i * 16 + hi * 8 + r;
                float v = acc[i][j].f[r] + bias;
                float g = 0.5f * v * (1.0f + erff(v * 0.70710678f));
                hbuf[(size_t)e * CAPn * Fn + (size_t)row * Fn + col] = f2bf(g);
            }
        }
    }
}

// ---------------------------------------------------------------------------
// Kernel 5: out = h @ w2[e] + b2[e]; weighted scatter-combine into y.
// ---------------------------------------------------------------------------
__global__ __launch_bounds__(256) void moe_gemm2(const unsigned short* __restrict__ hbuf,
                                                 const float* __restrict__ w2,
                                                 const float* __restrict__ b2,
                                                 const int* __restrict__ token_idx,
                                                 const float* __restrict__ token_w,
                                                 float* __restrict__ y) {
    __shared__ unsigned short sA[2][BM * LDA];
    __shared__ unsigned short sB[2][BK * LDB];

    const int e  = blockIdx.z;
    const int n0 = blockIdx.x * BN;
    const int m0 = blockIdx.y * BM;
    const int lane = threadIdx.x & 31;
    const int wave = threadIdx.x >> 5;
    const int wm = (wave >> 2) * 64;
    const int wn = (wave & 3) * 32;
    const int l  = lane & 15;
    const int hi = lane >> 4;

    AccT acc[4][2];
    #pragma unroll
    for (int i = 0; i < 4; ++i)
        #pragma unroll
        for (int j = 0; j < 2; ++j)
            acc[i][j].v = (v8f){0.f, 0.f, 0.f, 0.f, 0.f, 0.f, 0.f, 0.f};

    const unsigned short* Ae = hbuf + (size_t)e * CAPn * Fn;
    const float* Be = w2 + (size_t)e * Fn * Dn;
    gemm_core<Fn, Dn>(Ae, Be, m0, n0, sA, sB, acc);

    // epilogue: bias, then y[t, col] += route_w * out  (global_atomic_add_f32)
    #pragma unroll
    for (int i = 0; i < 4; ++i) {
        #pragma unroll
        for (int j = 0; j < 2; ++j) {
            int col = n0 + wn + j * 16 + l;
            float bias = b2[e * Dn + col];
            #pragma unroll
            for (int r = 0; r < 8; ++r) {
                int row = m0 + wm + i * 16 + hi * 8 + r;   // slot index c
                int t = token_idx[e * CAPn + row];
                if (t >= 0) {
                    float wt = token_w[e * CAPn + row];
                    atomicAdd(&y[(size_t)t * Dn + col], wt * (acc[i][j].f[r] + bias));
                }
            }
        }
    }
}

// ---------------------------------------------------------------------------
// Launch
// ---------------------------------------------------------------------------
extern "C" void kernel_launch(void* const* d_in, const int* in_sizes, int n_in,
                              void* d_out, int out_size, void* d_ws, size_t ws_size,
                              hipStream_t stream) {
    (void)in_sizes; (void)n_in; (void)out_size; (void)ws_size;

    const float*         x    = (const float*)d_in[0];
    const unsigned char* mask = (const unsigned char*)d_in[1];
    const float*         rw   = (const float*)d_in[2];
    const float*         w1   = (const float*)d_in[3];
    const float*         b1   = (const float*)d_in[4];
    const float*         w2   = (const float*)d_in[5];
    const float*         b2   = (const float*)d_in[6];
    float*               y    = (float*)d_out;

    char* ws = (char*)d_ws;
    int*            cnt       = (int*)(ws + 0);
    int*            token_idx = (int*)(ws + 4096);
    float*          token_w   = (float*)(ws + 4096 + En * CAPn * 4);
    unsigned short* xs        = (unsigned short*)(ws + (1u << 20));        // 16 MB
    unsigned short* hbuf      = (unsigned short*)(ws + (32u << 20));       // 64 MB

    moe_init<<<(Tn * Dn + 255) / 256, 256, 0, stream>>>(cnt, token_idx, token_w, y);
    moe_route<<<(Tn + 255) / 256, 256, 0, stream>>>(mask, rw, cnt, token_idx, token_w);
    moe_gather<<<En * CAPn, 256, 0, stream>>>(x, token_idx, xs);

    dim3 g1(Fn / BN, CAPn / BM, En);
    moe_gemm1<<<g1, 256, 0, stream>>>(xs, w1, b1, hbuf);

    dim3 g2(Dn / BN, CAPn / BM, En);
    moe_gemm2<<<g2, 256, 0, stream>>>(hbuf, w2, b2, token_idx, token_w, y);
}
